// WindowAttention_41918880809049
// MI455X (gfx1250) — matrix operand order repaired and checked by reading, not compiled, not run on hardware
//
#include <hip/hip_runtime.h>

#define DIM      1024
#define HEADS    16
#define HEAD_DIM 64
#define BATCH    8
#define SEQ      1024
#define SCALE_Q  0.125f   // HEAD_DIM^-0.5

typedef __attribute__((ext_vector_type(16))) __bf16 v16bf;
typedef __attribute__((ext_vector_type(8)))  float  v8f;
typedef unsigned short u16b;
typedef unsigned int   u32;

union FragAB { v16bf v; u32 u[8]; };

// float -> bf16, round-to-nearest-even
__device__ inline u16b f2bf(float f) {
    union { float f; u32 u; } c; c.f = f;
    u32 u = c.u;
    u32 r = u + 0x7FFFu + ((u >> 16) & 1u);
    return (u16b)(r >> 16);
}

// A-fragment (16x32 bf16, MxK): lane holds row M = lane%16.
// Per ISA 7.12.2: lanes 0-15 get K={0..7,16..23}, lanes 16-31 K={8..15,24..31};
// dword v<4 -> row dword [4*half+v], v>=4 -> [8+4*half+(v-4)].
__device__ inline v16bf load_frag_a(const u16b* base, int ld) {
    const int lane = threadIdx.x & 31;
    const int idx = lane & 15, half = lane >> 4;
    const u32* d = (const u32*)(base + idx * ld);
    FragAB f;
#pragma unroll
    for (int v = 0; v < 4; ++v) f.u[v] = d[4 * half + v];
#pragma unroll
    for (int v = 4; v < 8; ++v) f.u[v] = d[8 + 4 * half + (v - 4)];
    return f.v;
}

// B-fragment (32x16 bf16, KxN): lane holds column N = lane%16; source matrix is
// laid out so each fragment column is a contiguous K-row in memory.
// lanes 0-15 hold K={0..7,16..23}, lanes 16-31 hold K={8..15,24..31} (same as A).
__device__ inline v16bf load_frag_b(const u16b* base, int ld) {
    const int lane = threadIdx.x & 31;
    const int idx = lane & 15, half = lane >> 4;
    const u32* d = (const u32*)(base + idx * ld);
    FragAB f;
#pragma unroll
    for (int v = 0; v < 8; ++v) f.u[v] = d[(v < 4) ? (4 * half + v) : (8 + 4 * half + (v - 4))];
    return f.v;
}

__device__ inline v8f wmma_bf16(v16bf a, v16bf b, v8f c) {
    return __builtin_amdgcn_wmma_f32_16x16x32_bf16(false, a, false, b,
                                                   (short)0, c, false, false);
}

// Fragment set for a 32x64 output tile k-step: 2 A tiles, 4 B tiles.
struct Frags { v16bf a[2]; v16bf b[4]; };

__device__ inline void load_frags(Frags& f, const u16b* arow, const u16b* wbase, int k0) {
#pragma unroll
    for (int i = 0; i < 2; ++i) f.a[i] = load_frag_a(arow + i * 16 * DIM + k0, DIM);
#pragma unroll
    for (int t = 0; t < 4; ++t) f.b[t] = load_frag_b(wbase + t * 16 * DIM + k0, DIM);
}

__device__ inline void do_wmma8(v8f acc[2][4], const Frags& f) {
#pragma unroll
    for (int t = 0; t < 4; ++t) {
        acc[0][t] = wmma_bf16(f.a[0], f.b[t], acc[0][t]);
        acc[1][t] = wmma_bf16(f.a[1], f.b[t], acc[1][t]);
    }
}

// ---------------- conversion kernels ----------------
__global__ void cast_f32_bf16(const float* __restrict__ in, u16b* __restrict__ out, int n) {
    int stride = gridDim.x * blockDim.x;
    for (int i = blockIdx.x * blockDim.x + threadIdx.x; i < n; i += stride)
        out[i] = f2bf(in[i]);
}

// in: [K, N] f32 (y = x@W layout), out: [N, K] bf16 (K contiguous)
__global__ void transpose_cast(const float* __restrict__ in, u16b* __restrict__ out,
                               int K, int N) {
    long total = (long)K * N;
    long stride = (long)gridDim.x * blockDim.x;
    for (long i = blockIdx.x * (long)blockDim.x + threadIdx.x; i < total; i += stride) {
        long n = i / K, k = i - n * K;
        out[i] = f2bf(in[k * (long)N + n]);
    }
}

// ---------------- QKV GEMM: [8192,1024] @ [1024,3072] + bias ----------------
// wave tile = 32 rows x 64 cols, double-buffered K loop; scatters into
// Q[B,H,N,D] (pre-scaled by D^-1/2), K[B,H,N,D], Vt[B,H,D,N].
__global__ void gemm_qkv(const u16b* __restrict__ xbf, const u16b* __restrict__ wt,
                         const float* __restrict__ bias,
                         u16b* __restrict__ qb, u16b* __restrict__ kb,
                         u16b* __restrict__ vtb) {
    const int wid  = (blockIdx.x * blockDim.x + threadIdx.x) >> 5;
    const int lane = threadIdx.x & 31;
    const int idx = lane & 15, half = lane >> 4;
    const int colStrip = wid % 48;   // 48 strips of 64 over N=3072
    const int rowTile  = wid / 48;   // 256 tiles of 32 over M=8192

    const u16b* arow  = xbf + rowTile * 32 * DIM;
    const u16b* wbase = wt + colStrip * 64 * DIM;

    v8f acc[2][4] = {};
    Frags f0, f1;
    load_frags(f0, arow, wbase, 0);
    // ping-pong: issue next-set loads before consuming current set so the
    // compiler emits partial s_wait_loadcnt (overlap loads with WMMA).
#pragma unroll 1
    for (int k0 = 0; k0 < DIM - 64; k0 += 64) {
        load_frags(f1, arow, wbase, k0 + 32);
        do_wmma8(acc, f0);
        load_frags(f0, arow, wbase, k0 + 64);
        do_wmma8(acc, f1);
    }
    load_frags(f1, arow, wbase, DIM - 32);
    do_wmma8(acc, f0);   // k = DIM-64
    do_wmma8(acc, f1);   // k = DIM-32

#pragma unroll
    for (int rt = 0; rt < 2; ++rt) {
        const int rowBase = rowTile * 32 + rt * 16 + 8 * half;
#pragma unroll
        for (int t = 0; t < 4; ++t) {
            const int j = colStrip * 64 + t * 16 + idx;   // 0..3071
            const float bj = bias[j];
            const int sec = j >> 10;                       // 0=Q 1=K 2=V (uniform/wave)
            const int c = j & 1023;
            const int h = c >> 6, dd = c & 63;
#pragma unroll
            for (int r = 0; r < 8; ++r) {
                const int row = rowBase + r;               // 0..8191
                const int b_ = row >> 10, n = row & 1023;
                const float val = acc[rt][t][r] + bj;
                if (sec == 0)
                    qb[((b_ * HEADS + h) * SEQ + n) * HEAD_DIM + dd] = f2bf(val * SCALE_Q);
                else if (sec == 1)
                    kb[((b_ * HEADS + h) * SEQ + n) * HEAD_DIM + dd] = f2bf(val);
                else
                    vtb[((b_ * HEADS + h) * HEAD_DIM + dd) * SEQ + n] = f2bf(val);
            }
        }
    }
}

// ---------------- flash attention: one wave per (b,h,16-query tile) ----------------
__global__ void attn_kernel(const u16b* __restrict__ qb, const u16b* __restrict__ kb,
                            const u16b* __restrict__ vtb, u16b* __restrict__ aout) {
    __shared__ u16b lds[8][16 * 32];
    const int warp = threadIdx.x >> 5;
    const int wid  = blockIdx.x * 8 + warp;            // 0..8191
    const int lane = threadIdx.x & 31;
    const int idx = lane & 15, half = lane >> 4;
    const int b_ = wid >> 10;
    const int h  = (wid >> 6) & 15;
    const int qt = wid & 63;

    const u16b* Qp = qb + ((b_ * HEADS + h) * SEQ + qt * 16) * HEAD_DIM;
    const u16b* Kp = kb + (b_ * HEADS + h) * SEQ * HEAD_DIM;
    const u16b* Vp = vtb + (b_ * HEADS + h) * HEAD_DIM * SEQ;

    const v16bf aq0 = load_frag_a(Qp, HEAD_DIM);
    const v16bf aq1 = load_frag_a(Qp + 32, HEAD_DIM);

    float mrow[8], lrow[8];
    v8f o[4] = {};
#pragma unroll
    for (int r = 0; r < 8; ++r) { mrow[r] = -3.0e38f; lrow[r] = 0.f; }

    u16b* pl = &lds[warp][0];

    for (int m0 = 0; m0 < SEQ; m0 += 32) {
        v8f s0 = {}, s1 = {};
        s0 = wmma_bf16(aq0, load_frag_b(Kp + m0 * HEAD_DIM, HEAD_DIM), s0);
        s0 = wmma_bf16(aq1, load_frag_b(Kp + m0 * HEAD_DIM + 32, HEAD_DIM), s0);
        s1 = wmma_bf16(aq0, load_frag_b(Kp + (m0 + 16) * HEAD_DIM, HEAD_DIM), s1);
        s1 = wmma_bf16(aq1, load_frag_b(Kp + (m0 + 16) * HEAD_DIM + 32, HEAD_DIM), s1);

        // Issue V-fragment loads now; the softmax VALU chain below hides their latency.
        v16bf bv[4];
#pragma unroll
        for (int t = 0; t < 4; ++t)
            bv[t] = load_frag_b(Vp + (t * 16) * SEQ + m0, SEQ);

        // online softmax over 32-key block (row = r + 8*half, cols across 16 lanes)
#pragma unroll
        for (int r = 0; r < 8; ++r) {
            float bm = fmaxf(s0[r], s1[r]);
            bm = fmaxf(bm, __shfl_xor(bm, 1, 32));
            bm = fmaxf(bm, __shfl_xor(bm, 2, 32));
            bm = fmaxf(bm, __shfl_xor(bm, 4, 32));
            bm = fmaxf(bm, __shfl_xor(bm, 8, 32));
            const float mnew  = fmaxf(mrow[r], bm);
            const float alpha = __expf(mrow[r] - mnew);
            const float p0 = __expf(s0[r] - mnew);
            const float p1 = __expf(s1[r] - mnew);
            float ps = p0 + p1;
            ps += __shfl_xor(ps, 1, 32);
            ps += __shfl_xor(ps, 2, 32);
            ps += __shfl_xor(ps, 4, 32);
            ps += __shfl_xor(ps, 8, 32);
            lrow[r] = lrow[r] * alpha + ps;
            mrow[r] = mnew;
#pragma unroll
            for (int t = 0; t < 4; ++t) o[t][r] *= alpha;
            const int rowl = r + 8 * half;
            pl[rowl * 32 + idx]      = f2bf(p0);
            pl[rowl * 32 + 16 + idx] = f2bf(p1);
        }
        // per-wave LDS round-trip to relayout P (C-layout -> A-fragment layout)
        asm volatile("s_wait_dscnt 0" ::: "memory");
        const v16bf pa = load_frag_a(pl, 32);
#pragma unroll
        for (int t = 0; t < 4; ++t)
            o[t] = wmma_bf16(pa, bv[t], o[t]);
    }

    const int nrow = qt * 16 + 8 * half;
#pragma unroll
    for (int t = 0; t < 4; ++t)
#pragma unroll
        for (int r = 0; r < 8; ++r) {
            const float val = o[t][r] / lrow[r];
            aout[(b_ * SEQ + nrow + r) * DIM + h * HEAD_DIM + t * 16 + idx] = f2bf(val);
        }
}

// ---------------- proj GEMM: [8192,1024] @ [1024,1024] + bias -> f32 out ----------------
__global__ void gemm_proj(const u16b* __restrict__ aout, const u16b* __restrict__ wt,
                          const float* __restrict__ bias, float* __restrict__ out) {
    const int wid  = (blockIdx.x * blockDim.x + threadIdx.x) >> 5;
    const int lane = threadIdx.x & 31;
    const int idx = lane & 15, half = lane >> 4;
    const int colStrip = wid & 15;   // 16 strips of 64 over N=1024
    const int rowTile  = wid >> 4;   // 256 tiles of 32

    const u16b* arow  = aout + rowTile * 32 * DIM;
    const u16b* wbase = wt + colStrip * 64 * DIM;

    v8f acc[2][4] = {};
    Frags f0, f1;
    load_frags(f0, arow, wbase, 0);
#pragma unroll 1
    for (int k0 = 0; k0 < DIM - 64; k0 += 64) {
        load_frags(f1, arow, wbase, k0 + 32);
        do_wmma8(acc, f0);
        load_frags(f0, arow, wbase, k0 + 64);
        do_wmma8(acc, f1);
    }
    load_frags(f1, arow, wbase, DIM - 32);
    do_wmma8(acc, f0);
    do_wmma8(acc, f1);

#pragma unroll
    for (int rt = 0; rt < 2; ++rt) {
        const int rowBase = rowTile * 32 + rt * 16 + 8 * half;
#pragma unroll
        for (int t = 0; t < 4; ++t) {
            const int j = colStrip * 64 + t * 16 + idx;
            const float bj = bias[j];
#pragma unroll
            for (int r = 0; r < 8; ++r)
                out[(rowBase + r) * DIM + j] = acc[rt][t][r] + bj;
        }
    }
}

extern "C" void kernel_launch(void* const* d_in, const int* in_sizes, int n_in,
                              void* d_out, int out_size, void* d_ws, size_t ws_size,
                              hipStream_t stream) {
    const float* x      = (const float*)d_in[0];
    const float* W_qkv  = (const float*)d_in[1];
    const float* b_qkv  = (const float*)d_in[2];
    const float* W_proj = (const float*)d_in[3];
    const float* b_proj = (const float*)d_in[4];
    float* out = (float*)d_out;

    char* ws = (char*)d_ws;
    size_t off = 0;
    auto alloc = [&](size_t bytes) -> void* {
        void* p = ws + off;
        off += (bytes + 255) & ~(size_t)255;
        return p;
    };
    u16b* xbf    = (u16b*)alloc((size_t)8192 * 1024 * 2);   // x as bf16
    u16b* wqkvT  = (u16b*)alloc((size_t)3072 * 1024 * 2);   // W_qkv^T bf16
    u16b* wprojT = (u16b*)alloc((size_t)1024 * 1024 * 2);   // W_proj^T bf16
    u16b* qb     = (u16b*)alloc((size_t)BATCH * HEADS * SEQ * HEAD_DIM * 2);
    u16b* kb     = (u16b*)alloc((size_t)BATCH * HEADS * SEQ * HEAD_DIM * 2);
    u16b* vtb    = (u16b*)alloc((size_t)BATCH * HEADS * HEAD_DIM * SEQ * 2);
    u16b* aouts  = (u16b*)alloc((size_t)8192 * 1024 * 2);   // attention output bf16

    cast_f32_bf16<<<2048, 256, 0, stream>>>(x, xbf, 8192 * 1024);
    transpose_cast<<<2048, 256, 0, stream>>>(W_qkv, wqkvT, 1024, 3072);
    transpose_cast<<<1024, 256, 0, stream>>>(W_proj, wprojT, 1024, 1024);
    // 12288 waves (256 row tiles x 48 col strips), 8 waves/block
    gemm_qkv<<<1536, 256, 0, stream>>>(xbf, wqkvT, b_qkv, qb, kb, vtb);
    // 8192 waves (8*16*64 query tiles)
    attn_kernel<<<1024, 256, 0, stream>>>(qb, kb, vtb, aouts);
    // 4096 waves (256 row tiles x 16 col strips)
    gemm_proj<<<512, 256, 0, stream>>>(aouts, wprojT, b_proj, out);
}
